// HeteroResonance_24970939859672
// MI455X (gfx1250) — compile-verified, matrix operand order checked by reading, not run on hardware
//
#include <hip/hip_runtime.h>
#include <hip/hip_bf16.h>

#define B_   4
#define T_   4096
#define D_   2048
#define BOT_ 256
#define BT_  (B_ * T_)

typedef __attribute__((ext_vector_type(16))) __bf16 v16bf;
typedef __attribute__((ext_vector_type(8)))  float  v8f;

// ---------------------------------------------------------------------------
// Kernel 1: hdc f32 [BT, D] -> bf16 transposed HbT [D, BT] (LDS tile transpose)
// ---------------------------------------------------------------------------
__global__ __launch_bounds__(256)
void cvtT_kernel(const float* __restrict__ src, __bf16* __restrict__ dstT) {
    __shared__ __bf16 tile[32][33];
    const int r = blockIdx.x * 32;          // row tile in BT (512 tiles)
    const int c = blockIdx.y * 32;          // col tile in D  (64 tiles)
    const int tx = threadIdx.x & 31;
    const int ty = threadIdx.x >> 5;        // 8 row-groups
    #pragma unroll
    for (int rr = ty; rr < 32; rr += 8)
        tile[rr][tx] = (__bf16)src[(size_t)(r + rr) * D_ + c + tx];
    __syncthreads();
    #pragma unroll
    for (int rr = ty; rr < 32; rr += 8)
        dstT[(size_t)(c + rr) * BT_ + r + tx] = tile[tx][rr];
}

// ---------------------------------------------------------------------------
// Kernel 2: Q/K projection.  One wave -> one 16x16 bf16 output tile.
//   A = hdc rows (16xK), B = W^T (KxN) loaded from W rows. K-loop over D.
// ---------------------------------------------------------------------------
__global__ __launch_bounds__(128)
void proj_qk_kernel(const float* __restrict__ hdc,
                    const float* __restrict__ Wq, const float* __restrict__ bq,
                    const float* __restrict__ Wk, const float* __restrict__ bk,
                    __bf16* __restrict__ Qb, __bf16* __restrict__ Kb) {
    const int lane  = threadIdx.x & 31;
    const int wv    = threadIdx.x >> 5;
    const int mlo   = lane & 15;
    const int khalf = (lane >> 4) * 16;
    const int row0  = blockIdx.x * 16;                  // 0..BT step 16
    const int n0    = (blockIdx.y * 4 + wv) * 16;       // 0..BOT-16
    const bool isK  = (blockIdx.z != 0);
    const float* W    = isK ? Wk : Wq;
    const float* bias = isK ? bk : bq;
    __bf16*      dst  = isK ? Kb : Qb;

    const float* ap = hdc + (size_t)(row0 + mlo) * D_ + khalf;
    const float* bp = W   + (size_t)(n0  + mlo) * D_ + khalf;

    v8f acc = {};
    for (int k0 = 0; k0 < D_; k0 += 32) {
        v16bf a, b;
        #pragma unroll
        for (int i = 0; i < 16; ++i) {
            a[i] = (__bf16)ap[k0 + i];
            b[i] = (__bf16)bp[k0 + i];
        }
        acc = __builtin_amdgcn_wmma_f32_16x16x32_bf16(false, a, false, b,
                                                      (short)0, acc, false, false);
    }
    const float bv = bias[n0 + mlo];
    #pragma unroll
    for (int r = 0; r < 8; ++r) {
        const int m = r + 8 * (lane >> 4);              // C layout: M = r + 8*hi
        dst[(size_t)(row0 + m) * BOT_ + n0 + mlo] = (__bf16)(acc[r] + bv);
    }
}

// ---------------------------------------------------------------------------
// Kernel 3: masked-softmax row statistics (online max / sum), WMMA scores.
//   One wave per 16-row strip.  j iterates only over the causal band.
// ---------------------------------------------------------------------------
__global__ __launch_bounds__(128)
void softmax_stats_kernel(const __bf16* __restrict__ Qb, const __bf16* __restrict__ Kb,
                          float* __restrict__ mbuf, float* __restrict__ lbuf) {
    const int lane  = threadIdx.x & 31;
    const int wv    = threadIdx.x >> 5;
    const int mlo   = lane & 15;
    const int hi    = lane >> 4;
    const int khalf = hi * 16;
    const int rowTile = blockIdx.x * 4 + wv;            // 0..1023
    const int bIdx    = rowTile >> 8;                   // 256 strips per batch
    const int r0      = (rowTile & 255) * 16;
    const size_t base = (size_t)bIdx * T_ * BOT_;
    const float scale = 0.0625f;                        // BOT^-0.5

    // Preload all 8 Q fragments for this strip (L2-resident).
    v16bf qf[8];
    #pragma unroll
    for (int kk = 0; kk < 8; ++kk)
        qf[kk] = *(const v16bf*)(Qb + base + (size_t)(r0 + mlo) * BOT_ + kk * 32 + khalf);

    float mrun[8], lrun[8];
    #pragma unroll
    for (int r = 0; r < 8; ++r) { mrun[r] = -3.0e38f; lrun[r] = 0.0f; }

    const int jTileMax = (r0 + 11) >> 4;                // inclusive (j <= i-4 band)
    for (int jt = 0; jt <= jTileMax; ++jt) {
        const int j0 = jt * 16;
        v8f s = {};
        #pragma unroll
        for (int kk = 0; kk < 8; ++kk) {
            v16bf kb = *(const v16bf*)(Kb + base + (size_t)(j0 + mlo) * BOT_ + kk * 32 + khalf);
            s = __builtin_amdgcn_wmma_f32_16x16x32_bf16(false, qf[kk], false, kb,
                                                        (short)0, s, false, false);
        }
        #pragma unroll
        for (int r = 0; r < 8; ++r) {
            const int i = r0 + r + 8 * hi;
            const int j = j0 + mlo;
            float v = (j <= i - 4) ? s[r] * scale : -65000.0f;
            float mx = v;
            #pragma unroll
            for (int o = 1; o < 16; o <<= 1) mx = fmaxf(mx, __shfl_xor(mx, o, 16));
            const float mnew = fmaxf(mrun[r], mx);
            float e = __expf(v - mnew);
            #pragma unroll
            for (int o = 1; o < 16; o <<= 1) e += __shfl_xor(e, o, 16);
            lrun[r] = lrun[r] * __expf(mrun[r] - mnew) + e;
            mrun[r] = mnew;
        }
    }
    if (mlo == 0) {
        #pragma unroll
        for (int r = 0; r < 8; ++r) {
            const int i = r0 + r + 8 * hi;
            mbuf[(size_t)bIdx * T_ + i] = mrun[r];
            lbuf[(size_t)bIdx * T_ + i] = lrun[r];
        }
    }
}

// ---------------------------------------------------------------------------
// Kernel 4: fused  out = (0.7*decay + 0.3*softmax(qk^T)) @ hdc.
//   One block per 16-row strip, 8 waves x 256 D-columns (16 acc tiles each).
//   Waves 0-1 produce the combined 16x32 bf16 tile per j-step into LDS
//   (double buffered, one barrier per step); all waves consume it as the
//   WMMA A-fragment against L2-resident transposed hdc.
// ---------------------------------------------------------------------------
#define NT 16
__global__ __launch_bounds__(256)
void fused_out_kernel(const float* __restrict__ decay,
                      const __bf16* __restrict__ Qb, const __bf16* __restrict__ Kb,
                      const __bf16* __restrict__ HbT,
                      const float* __restrict__ mbuf, const float* __restrict__ lbuf,
                      float* __restrict__ out) {
    __shared__ alignas(32) __bf16 pl[2][16 * 32];
    const int lane  = threadIdx.x & 31;
    const int wv    = threadIdx.x >> 5;
    const int mlo   = lane & 15;
    const int hi    = lane >> 4;
    const int khalf = hi * 16;
    const int rowTile = blockIdx.x;                     // 0..1023
    const int bIdx    = rowTile >> 8;
    const int r0      = (rowTile & 255) * 16;
    const size_t qkBase  = (size_t)bIdx * T_ * BOT_;
    const size_t decBase = ((size_t)bIdx * T_ + r0) * T_;
    const float scale = 0.0625f;

    // Producer-only per-row softmax stats.
    float mrow[8], inv3[8];
    if (wv < 2) {
        #pragma unroll
        for (int r = 0; r < 8; ++r) {
            const int i = r0 + r + 8 * hi;
            mrow[r] = mbuf[(size_t)bIdx * T_ + i];
            inv3[r] = 0.3f / lbuf[(size_t)bIdx * T_ + i];
        }
    }

    v8f acc[NT] = {};
    const int col0 = wv * 256;
    const int jAllowedMax = r0 + 11;
    const int nSteps = T_ / 32;                         // 128

    for (int step = 0; step < nSteps; ++step) {
        const int j0  = step * 32;
        const int buf = step & 1;
        if (wv < 2) {
            const int js = j0 + wv * 16;
            const bool doScore = (js <= jAllowedMax);
            v8f sc = {};
            if (doScore) {
                #pragma unroll
                for (int kk = 0; kk < 8; ++kk) {
                    v16bf qa = *(const v16bf*)(Qb + qkBase + (size_t)(r0 + mlo) * BOT_ + kk * 32 + khalf);
                    v16bf kb = *(const v16bf*)(Kb + qkBase + (size_t)(js + mlo) * BOT_ + kk * 32 + khalf);
                    sc = __builtin_amdgcn_wmma_f32_16x16x32_bf16(false, qa, false, kb,
                                                                 (short)0, sc, false, false);
                }
            }
            #pragma unroll
            for (int r = 0; r < 8; ++r) {
                const int i = r + 8 * hi;               // local row
                const int j = js + mlo;
                float v = 0.7f * decay[decBase + (size_t)i * T_ + j];
                if (doScore && (j <= r0 + i - 4))
                    v += __expf(sc[r] * scale - mrow[r]) * inv3[r];
                pl[buf][i * 32 + wv * 16 + mlo] = (__bf16)v;
            }
        }
        __syncthreads();
        // Consume: A = combined tile (16x32), B = HbT columns (contiguous b128).
        const v16bf a = *(const v16bf*)(&pl[buf][mlo * 32 + khalf]);
        const __bf16* hb = HbT + (size_t)col0 * BT_ + (size_t)bIdx * T_ + j0 + khalf;
        #pragma unroll
        for (int t = 0; t < NT; ++t) {
            v16bf bb = *(const v16bf*)(hb + (size_t)(t * 16 + mlo) * BT_);
            acc[t] = __builtin_amdgcn_wmma_f32_16x16x32_bf16(false, a, false, bb,
                                                             (short)0, acc[t], false, false);
        }
    }

    const size_t outBase = ((size_t)bIdx * T_ + r0) * D_ + col0;
    #pragma unroll
    for (int t = 0; t < NT; ++t) {
        #pragma unroll
        for (int r = 0; r < 8; ++r)
            out[outBase + (size_t)(r + 8 * hi) * D_ + t * 16 + mlo] = acc[t][r];
    }
}

// ---------------------------------------------------------------------------
extern "C" void kernel_launch(void* const* d_in, const int* in_sizes, int n_in,
                              void* d_out, int out_size, void* d_ws, size_t ws_size,
                              hipStream_t stream) {
    const float* hdc   = (const float*)d_in[0];   // [B,T,D]
    const float* decay = (const float*)d_in[1];   // [B,T,T]
    const float* Wq    = (const float*)d_in[2];   // [BOT,D]
    const float* bq    = (const float*)d_in[3];   // [BOT]
    const float* Wk    = (const float*)d_in[4];   // [BOT,D]
    const float* bk    = (const float*)d_in[5];   // [BOT]
    float* out = (float*)d_out;

    char* ws = (char*)d_ws;
    const size_t szHbT = (size_t)D_ * BT_ * sizeof(__bf16);     // 64 MiB
    const size_t szQK  = (size_t)BT_ * BOT_ * sizeof(__bf16);   // 8 MiB each
    __bf16* HbT = (__bf16*)ws;
    __bf16* Qb  = (__bf16*)(ws + szHbT);
    __bf16* Kb  = (__bf16*)(ws + szHbT + szQK);
    float*  mbf = (float*)(ws + szHbT + 2 * szQK);
    float*  lbf = mbf + BT_;

    cvtT_kernel<<<dim3(BT_ / 32, D_ / 32), 256, 0, stream>>>(hdc, HbT);
    proj_qk_kernel<<<dim3(BT_ / 16, BOT_ / 64, 2), 128, 0, stream>>>(hdc, Wq, bq, Wk, bk, Qb, Kb);
    softmax_stats_kernel<<<BT_ / 64, 128, 0, stream>>>(Qb, Kb, mbf, lbf);
    fused_out_kernel<<<BT_ / 16, 256, 0, stream>>>(decay, Qb, Kb, HbT, mbf, lbf, out);
}
